// MSIB_GIN_57724360458775
// MI455X (gfx1250) — matrix operand order repaired and checked by reading, not compiled, not run on hardware
//
#include <hip/hip_runtime.h>
#include <hip/hip_bf16.h>

typedef __attribute__((ext_vector_type(16))) _Float16 v16h;
typedef __attribute__((ext_vector_type(8)))  _Float16 v8h;
typedef __attribute__((ext_vector_type(4)))  _Float16 v4h;
typedef __attribute__((ext_vector_type(8)))  float    v8f;

#define GIN_D 128
#define GIN_G 256

// ---------------- segment max (batch sorted, node_imp >= 0) ----------------
__global__ void gin_zero_kernel(float* p, int n) {
    int i = blockIdx.x * 256 + threadIdx.x;
    if (i < n) p[i] = 0.0f;
}

__global__ void gin_segmax_kernel(const float* __restrict__ imp,
                                  const int* __restrict__ batch,
                                  float* __restrict__ segmax, int n) {
    int i = blockIdx.x * 256 + threadIdx.x;
    if (i < n) {
        // node_imp in [0,1): uint bit-pattern compare == float compare
        atomicMax((unsigned int*)&segmax[batch[i]], __float_as_uint(imp[i]));
    }
}

// ---------------- x * importance scaling ----------------
__global__ void gin_scale_kernel(const float* __restrict__ x,
                                 const float* __restrict__ imp,
                                 const int* __restrict__ batch,
                                 const float* __restrict__ segmax,
                                 float* __restrict__ xs, int n) {
    int i = blockIdx.x * 256 + threadIdx.x;
    if (i >= n * GIN_D) return;
    int r = i >> 7;
    float ni = imp[r] / (segmax[batch[r]] + 1e-10f);
    ni = (2.0f * ni - 1.0f) * (1.0f / 40.0f) + 1.0f;   // (2ni-1)/(2*20)+1
    xs[i] = x[i] * ni;
}

// ---------------- weight convert f32 -> f16, transposed [n*128+k] ----------------
__global__ void gin_wconv_kernel(const float* __restrict__ W1,
                                 const float* __restrict__ W2,
                                 _Float16* __restrict__ wt1,
                                 _Float16* __restrict__ wt2) {
    int i = blockIdx.x * 256 + threadIdx.x;   // 16384 = 128x128
    if (i >= GIN_D * GIN_D) return;
    int k = i >> 7;
    int n = i & 127;
    wt1[n * GIN_D + k] = (_Float16)W1[i];
    wt2[n * GIN_D + k] = (_Float16)W2[i];
}

// ---------------- agg = 0 (float4 fill) ----------------
__global__ void gin_zeroagg_kernel(float4* __restrict__ agg, int n4) {
    int i = blockIdx.x * 256 + threadIdx.x;
    if (i < n4) agg[i] = make_float4(0.f, 0.f, 0.f, 0.f);
}

// ---------------- scatter-add: agg[dst] += x[src]; one wave32 per edge ----------------
__global__ __launch_bounds__(256) void gin_scatter_kernel(
    const float* __restrict__ x, int stride, int coloff,
    const int* __restrict__ ei, float* __restrict__ agg, int e_cnt) {
    int e = blockIdx.x * 8 + (threadIdx.x >> 5);
    if (e >= e_cnt) return;
    int lane = threadIdx.x & 31;
    int src = ei[e];
    int dst = ei[e_cnt + e];
    float4 v = *(const float4*)&x[(size_t)src * stride + coloff + lane * 4];
    float* hp = &agg[(size_t)dst * GIN_D + lane * 4];
    atomicAdd(hp + 0, v.x);
    atomicAdd(hp + 1, v.y);
    atomicAdd(hp + 2, v.z);
    atomicAdd(hp + 3, v.w);
}

// ---------------- fused MLP: out = relu(relu((x+agg)@W1+b1)@W2+b2), WMMA f16 --------
// 128 rows per block, 8 waves, 16-row stripe per wave, K=128 in 4 WMMA steps.
// Dynamic LDS: lds_h (32KB) | lds_w1 (32KB) | lds_w2 (32KB) = 96KB.
__global__ __launch_bounds__(256) void gin_mlp_wmma_kernel(
    const float* __restrict__ xprev,      // node feats, given stride/coloff
    int stride, int coloff_in,
    const float* __restrict__ agg,        // N x 128
    const _Float16* __restrict__ w1t,     // [n*128+k] f16
    const float* __restrict__ b1,
    const _Float16* __restrict__ w2t,
    const float* __restrict__ b2,
    float* __restrict__ xout,             // node-out base, stride 384
    int n_rows, int col_off) {
    extern __shared__ _Float16 smem[];
    _Float16* lds_h  = smem;                       // 16384 halfs
    _Float16* lds_w1 = smem + GIN_D * GIN_D;       // 16384 halfs
    _Float16* lds_w2 = smem + 2 * GIN_D * GIN_D;   // 16384 halfs

    const int tid  = threadIdx.x;
    const int wave = tid >> 5;
    const int lane = tid & 31;
    const int n15  = lane & 15;
    const int hi   = lane >> 4;
    const int rowBase = blockIdx.x * 128;
    const int R = wave * 16;                       // local row base of this wave

    // async-stage W1^T and W2^T into LDS (ASYNCcnt path, no VGPR round-trip)
    {
        unsigned l1 = (unsigned)(size_t)lds_w1;
        unsigned l2 = (unsigned)(size_t)lds_w2;
        unsigned long long g1 = (unsigned long long)(size_t)w1t;
        unsigned long long g2 = (unsigned long long)(size_t)w2t;
        for (int i = tid; i < 2048; i += 256) {    // 2048 x 16B = 32KB each
            unsigned goff = (unsigned)i * 16u;
            unsigned d1 = l1 + goff;
            unsigned d2 = l2 + goff;
            asm volatile("global_load_async_to_lds_b128 %0, %1, %2"
                         :: "v"(d1), "v"(goff), "s"(g1) : "memory");
            asm volatile("global_load_async_to_lds_b128 %0, %1, %2"
                         :: "v"(d2), "v"(goff), "s"(g2) : "memory");
        }
    }
    // stage H = x + agg, f32 -> f16 (overlaps with async weight staging)
    for (int i = tid; i < 4096; i += 256) {        // 4096 float4 = 128x128
        int r  = i >> 5;
        int c4 = (i & 31) << 2;
        int grow = rowBase + r;
        float4 v = {0.0f, 0.0f, 0.0f, 0.0f};
        if (grow < n_rows) {
            float4 xv = *(const float4*)&xprev[(size_t)grow * stride + coloff_in + c4];
            float4 av = *(const float4*)&agg[(size_t)grow * GIN_D + c4];
            v.x = xv.x + av.x; v.y = xv.y + av.y;
            v.z = xv.z + av.z; v.w = xv.w + av.w;
        }
        v4h o = {(_Float16)v.x, (_Float16)v.y, (_Float16)v.z, (_Float16)v.w};
        *(v4h*)&lds_h[r * GIN_D + c4] = o;
    }
    asm volatile("s_wait_asynccnt 0" ::: "memory");
    __syncthreads();   // the only block-wide sync

    const int arow = R + n15;                      // A-operand row for this lane
    v16h a[4];
    v8f acc[8];

    // ---- GEMM1: o1 = H @ W1 + b1 ----
#pragma unroll
    for (int kb = 0; kb < 4; ++kb) {
        int k0 = kb * 32 + hi * 16;
        v8h lo  = *(const v8h*)&lds_h[arow * GIN_D + k0];
        v8h hi8 = *(const v8h*)&lds_h[arow * GIN_D + k0 + 8];
        a[kb] = __builtin_shufflevector(lo, hi8, 0,1,2,3,4,5,6,7,8,9,10,11,12,13,14,15);
    }
#pragma unroll
    for (int c = 0; c < 8; ++c) {
        float bv = b1[c * 16 + n15];
#pragma unroll
        for (int i = 0; i < 8; ++i) acc[c][i] = bv;
#pragma unroll
        for (int kb = 0; kb < 4; ++kb) {
            v16h b = *(const v16h*)&lds_w1[(c * 16 + n15) * GIN_D + kb * 32 + hi * 16];
            acc[c] = __builtin_amdgcn_wmma_f32_16x16x32_f16(
                false, a[kb], false, b, (short)0, acc[c], false, false);
        }
    }

    // write relu(o1) back into lds_h: wave-private 16-row stripe, no block sync needed
#pragma unroll
    for (int c = 0; c < 8; ++c) {
#pragma unroll
        for (int i = 0; i < 8; ++i) {
            float v = acc[c][i];
            v = v > 0.0f ? v : 0.0f;
            lds_h[(R + i + hi * 8) * GIN_D + c * 16 + n15] = (_Float16)v;
        }
    }

    // ---- GEMM2: x = relu(o1 @ W2 + b2) ----
#pragma unroll
    for (int kb = 0; kb < 4; ++kb) {
        int k0 = kb * 32 + hi * 16;
        v8h lo  = *(const v8h*)&lds_h[arow * GIN_D + k0];
        v8h hi8 = *(const v8h*)&lds_h[arow * GIN_D + k0 + 8];
        a[kb] = __builtin_shufflevector(lo, hi8, 0,1,2,3,4,5,6,7,8,9,10,11,12,13,14,15);
    }
#pragma unroll
    for (int c = 0; c < 8; ++c) {
        float bv = b2[c * 16 + n15];
#pragma unroll
        for (int i = 0; i < 8; ++i) acc[c][i] = bv;
#pragma unroll
        for (int kb = 0; kb < 4; ++kb) {
            v16h b = *(const v16h*)&lds_w2[(c * 16 + n15) * GIN_D + kb * 32 + hi * 16];
            acc[c] = __builtin_amdgcn_wmma_f32_16x16x32_f16(
                false, a[kb], false, b, (short)0, acc[c], false, false);
        }
    }
    // store relu to global node output (stride 384, offset col_off)
#pragma unroll
    for (int c = 0; c < 8; ++c) {
#pragma unroll
        for (int i = 0; i < 8; ++i) {
            int grow = rowBase + R + i + hi * 8;
            if (grow < n_rows) {
                float v = acc[c][i];
                v = v > 0.0f ? v : 0.0f;
                xout[(size_t)grow * 384 + col_off + c * 16 + n15] = v;
            }
        }
    }
}

// ---------------- graph pooling: segment_sum over sorted batch ----------------
__global__ void gin_pool_kernel(const float* __restrict__ nodes,   // N x 384
                                const int* __restrict__ batch,
                                float* __restrict__ gout, int n) {
    int g   = blockIdx.x;
    int col = blockIdx.y * 128 + threadIdx.x;
    // lower_bound(batch, g)
    int lo = 0, hi = n;
    while (lo < hi) { int m = (lo + hi) >> 1; if (batch[m] < g) lo = m + 1; else hi = m; }
    int s = lo;
    lo = 0; hi = n;
    while (lo < hi) { int m = (lo + hi) >> 1; if (batch[m] < g + 1) lo = m + 1; else hi = m; }
    int e = lo;
    float sum = 0.0f;
    for (int r = s; r < e; ++r) sum += nodes[(size_t)r * 384 + col];
    gout[(size_t)g * 384 + col] = sum;
}

extern "C" void kernel_launch(void* const* d_in, const int* in_sizes, int n_in,
                              void* d_out, int out_size, void* d_ws, size_t ws_size,
                              hipStream_t stream) {
    const float* x        = (const float*)d_in[0];
    const int*   edge     = (const int*)d_in[1];
    const int*   batch    = (const int*)d_in[2];
    const float* node_imp = (const float*)d_in[3];

    const int N = in_sizes[3];
    const int E = in_sizes[1] / 2;

    // workspace layout
    float*    segmax = (float*)d_ws;
    float*    xs     = segmax + 256;                 // N*128 f32 (scaled input)
    float*    agg    = xs + (size_t)N * GIN_D;       // N*128 f32 (neighbor sum)
    _Float16* wt1    = (_Float16*)(agg + (size_t)N * GIN_D);
    _Float16* wt2    = wt1 + GIN_D * GIN_D;

    float* graph_emb = (float*)d_out;                        // 256 x 384
    float* nodes_out = (float*)d_out + (size_t)GIN_G * 384;  // N x 384

    // importance normalization + input scaling
    gin_zero_kernel<<<1, 256, 0, stream>>>(segmax, GIN_G);
    gin_segmax_kernel<<<(N + 255) / 256, 256, 0, stream>>>(node_imp, batch, segmax, N);
    gin_scale_kernel<<<(N * GIN_D + 255) / 256, 256, 0, stream>>>(
        x, node_imp, batch, segmax, xs, N);

    const float* xprev = xs;
    int stride = GIN_D, coloff_in = 0;
    const int lds_bytes = 3 * GIN_D * GIN_D * sizeof(_Float16);  // 96 KB

    for (int l = 0; l < 3; ++l) {
        const float* W1 = (const float*)d_in[4 + 4 * l + 0];
        const float* b1 = (const float*)d_in[4 + 4 * l + 1];
        const float* W2 = (const float*)d_in[4 + 4 * l + 2];
        const float* b2 = (const float*)d_in[4 + 4 * l + 3];

        gin_wconv_kernel<<<64, 256, 0, stream>>>(W1, W2, wt1, wt2);
        gin_zeroagg_kernel<<<(N * GIN_D / 4 + 255) / 256, 256, 0, stream>>>(
            (float4*)agg, N * GIN_D / 4);
        gin_scatter_kernel<<<(E + 7) / 8, 256, 0, stream>>>(
            xprev, stride, coloff_in, edge, agg, E);
        gin_mlp_wmma_kernel<<<(N + 127) / 128, 256, lds_bytes, stream>>>(
            xprev, stride, coloff_in, agg, wt1, b1, wt2, b2, nodes_out, N, l * GIN_D);

        xprev = nodes_out;
        stride = 384;
        coloff_in = l * GIN_D;
    }

    gin_pool_kernel<<<dim3(GIN_G, 3), 128, 0, stream>>>(nodes_out, batch, graph_emb, N);
}
